// Informer_66460323938535
// MI455X (gfx1250) — matrix operand order repaired and checked by reading, not compiled, not run on hardware
//
#include <hip/hip_runtime.h>
#include <hip/hip_bf16.h>

// Problem constants (fixed by the reference)
#define B_   4
#define L_   2048
#define E_   7
#define D_   512
#define H_   8
#define DH_  64
#define P_   720
#define NL_  2
#define U_   40
#define SCALE_ 0.125f   // 1/sqrt(64)

typedef __attribute__((ext_vector_type(16))) _Float16 v16h;
typedef __attribute__((ext_vector_type(8)))  _Float16 v8h;
typedef __attribute__((ext_vector_type(4)))  _Float16 v4h;
typedef __attribute__((ext_vector_type(8)))  float    v8f;

union HF16 { v16h v; v8h h[2]; };

// ---------------------------------------------------------------------------
// CDNA5 async global->LDS copy path (ASYNCcnt), guarded: falls back to
// register staging if the toolchain does not expose the builtins.
// Builtin signature (from hipcc diagnostic): params are pointers to
// int __attribute__((vector_size(16))) in AS1 (global) / AS3 (LDS).
// ---------------------------------------------------------------------------
#ifndef USE_ASYNC_LDS
#if defined(__has_builtin)
#if __has_builtin(__builtin_amdgcn_global_load_async_to_lds_b128) && \
    __has_builtin(__builtin_amdgcn_s_wait_asynccnt)
#define USE_ASYNC_LDS 1
#endif
#endif
#endif
#ifndef USE_ASYNC_LDS
#define USE_ASYNC_LDS 0
#endif

typedef int v4i_ __attribute__((vector_size(16)));
#if USE_ASYNC_LDS
typedef __attribute__((address_space(1))) v4i_ gv4i;   // global int4
typedef __attribute__((address_space(3))) v4i_ lv4i;   // LDS int4
#endif

__device__ __forceinline__ void copy16B(_Float16* dst, const _Float16* src) {
#if USE_ASYNC_LDS
    // global_load_async_to_lds_b128: 16 bytes per lane, no VGPR round-trip
    __builtin_amdgcn_global_load_async_to_lds_b128(
        (gv4i*)(unsigned long long)(uintptr_t)src,
        (lv4i*)(unsigned int)(uintptr_t)dst, 0, 0);
#else
    *(v4i_*)dst = *(const v4i_*)src;
#endif
}

__device__ __forceinline__ void wait_copies_keep4() {
#if USE_ASYNC_LDS
    __builtin_amdgcn_s_wait_asynccnt(4);   // wait current buffer, next stays in flight
#endif
}
__device__ __forceinline__ void wait_copies_all() {
#if USE_ASYNC_LDS
    __builtin_amdgcn_s_wait_asynccnt(0);
#endif
}

// ---------------------------------------------------------------------------
// Weight convert: Wt[n][k] = (f16) W[k][n]   (N-major so B-frags are contiguous)
// ---------------------------------------------------------------------------
__global__ void wcvt_kernel(const float* __restrict__ W, _Float16* __restrict__ Wt) {
    int gid = blockIdx.x * blockDim.x + threadIdx.x;   // D*D threads
    int n = gid / D_, k = gid % D_;
    Wt[(size_t)n * D_ + k] = (_Float16)W[(size_t)k * D_ + n];
}

// ---------------------------------------------------------------------------
// Activation convert fp32 -> f16 (4 elements / thread)
// ---------------------------------------------------------------------------
__global__ void cvt16_kernel(const float* __restrict__ src, _Float16* __restrict__ dst) {
    size_t gid = ((size_t)blockIdx.x * blockDim.x + threadIdx.x) * 4;
    float4 f = *(const float4*)(src + gid);
    v4h h;
    h[0] = (_Float16)f.x; h[1] = (_Float16)f.y;
    h[2] = (_Float16)f.z; h[3] = (_Float16)f.w;
    *(v4h*)(dst + gid) = h;
}

// ---------------------------------------------------------------------------
// Embedding: h[m,d] = sum_e x[m,e]*emb_w[e,d] + emb_b[d]   (K=7, VALU)
// ---------------------------------------------------------------------------
__global__ void embed_kernel(const float* __restrict__ x, const float* __restrict__ w,
                             const float* __restrict__ b, float* __restrict__ h) {
    size_t gid = (size_t)blockIdx.x * blockDim.x + threadIdx.x;
    if (gid >= (size_t)B_ * L_ * D_) return;
    int d = gid % D_; size_t m = gid / D_;
    float s = b[d];
#pragma unroll
    for (int e = 0; e < E_; e++) s += x[m * E_ + e] * w[e * D_ + d];
    h[gid] = s;
}

// ---------------------------------------------------------------------------
// WMMA GEMM: C[M,N] = A16[M,K](f16) * Bt[N,K](f16, col-major B) + bias
// MODE 0: fp32 row-major out (RELU optional).  MODE 1: f16 head-major out.
// 256 thr (8 waves, 4 along M x 2 along N). Tile 128x128x32, double-buffered
// LDS fed by async global->LDS copies.
// ---------------------------------------------------------------------------
template <int MODE, int RELU>
__global__ __launch_bounds__(256) void gemm_f16(
    const _Float16* __restrict__ A, const _Float16* __restrict__ Bt,
    const float* __restrict__ bias, float* __restrict__ C32,
    _Float16* __restrict__ C16, int M, int N, int K)
{
    __shared__ _Float16 As[2][128 * 40];   // 40-half row pitch -> 16B alignment
    __shared__ _Float16 Bs[2][128 * 40];
    const int tid  = threadIdx.x;
    const int lane = tid & 31, wave = tid >> 5;
    const int wm = wave & 3, wn = wave >> 2;
    const int l16 = lane & 15, hsel = lane >> 4;
    const int bM = blockIdx.y * 128, bN = blockIdx.x * 128;
    const int lr = tid >> 1, lc = (tid & 1) * 16;

    auto stage = [&](int buf, int k0) {   // 4 x 16B per thread = 4 async ops/wave
        const _Float16* asrc = A + (size_t)(bM + lr) * K + k0 + lc;
        _Float16* adst = &As[buf][lr * 40 + lc];
        copy16B(adst,     asrc);
        copy16B(adst + 8, asrc + 8);
        const _Float16* bsrc = Bt + (size_t)(bN + lr) * K + k0 + lc;
        _Float16* bdst = &Bs[buf][lr * 40 + lc];
        copy16B(bdst,     bsrc);
        copy16B(bdst + 8, bsrc + 8);
    };

    v8f acc[2][4] = {};
    const int nk = K >> 5;
    stage(0, 0);

    for (int kk = 0; kk < nk; kk++) {
        const int cur = kk & 1;
        if (kk + 1 < nk) { stage(cur ^ 1, (kk + 1) * 32); wait_copies_keep4(); }
        else             { wait_copies_all(); }
        __syncthreads();

        HF16 af[2], bf[4];
#pragma unroll
        for (int ms = 0; ms < 2; ms++) {
            // A-layout: lane<16 -> K{0..7,16..23}; lane>=16 -> K{8..15,24..31}
            const _Float16* ap = &As[cur][(wm * 32 + ms * 16 + l16) * 40 + hsel * 8];
            af[ms].h[0] = *(const v8h*)ap;
            af[ms].h[1] = *(const v8h*)(ap + 16);
        }
#pragma unroll
        for (int ns = 0; ns < 4; ns++) {
            // B-layout: lane<16 -> K{0..7,16..23}; lane>=16 -> K{8..15,24..31}
            const _Float16* bp = &Bs[cur][(wn * 64 + ns * 16 + l16) * 40 + hsel * 8];
            bf[ns].h[0] = *(const v8h*)bp;
            bf[ns].h[1] = *(const v8h*)(bp + 16);
        }
#pragma unroll
        for (int ms = 0; ms < 2; ms++)
#pragma unroll
            for (int ns = 0; ns < 4; ns++)
                acc[ms][ns] = __builtin_amdgcn_wmma_f32_16x16x32_f16(
                    false, af[ms].v, false, bf[ns].v, (short)0, acc[ms][ns], false, false);
        __syncthreads();
    }

#pragma unroll
    for (int ms = 0; ms < 2; ms++) {
#pragma unroll
        for (int ns = 0; ns < 4; ns++) {
            int col = bN + wn * 64 + ns * 16 + l16;
            float bv = bias[col];
#pragma unroll
            for (int i = 0; i < 8; i++) {
                int row = bM + wm * 32 + ms * 16 + hsel * 8 + i;
                float v = acc[ms][ns][i] + bv;
                if (RELU) v = fmaxf(v, 0.f);
                if (MODE == 0) {
                    C32[(size_t)row * N + col] = v;
                } else {
                    int b = row >> 11, l = row & (L_ - 1);
                    int hh = col >> 6, dh = col & (DH_ - 1);
                    C16[(((size_t)(b * H_ + hh)) * L_ + l) * DH_ + dh] = (_Float16)v;
                }
            }
        }
    }
}

// ---------------------------------------------------------------------------
// Streaming score statistics: M[bh,q] = max_k(score) - mean_k(score)
// WMMA over 128-query x 128-key chunks; scores never touch HBM.
// Double-buffered async K-chunk staging.
// ---------------------------------------------------------------------------
__global__ __launch_bounds__(256) void score_stats(
    const _Float16* __restrict__ q16, const _Float16* __restrict__ k16,
    float* __restrict__ Mstat)
{
    __shared__ _Float16 Qs[128 * 72];
    __shared__ _Float16 Ks[2][128 * 72];
    __shared__ float sMax[128][2];
    __shared__ float sSum[128][2];
    const int tid  = threadIdx.x;
    const int lane = tid & 31, wave = tid >> 5;
    const int wm = wave & 3, wn = wave >> 2;
    const int l16 = lane & 15, hsel = lane >> 4;
    const int bh = blockIdx.y;
    const int qbase = blockIdx.x * 128;
    const _Float16* qptr = q16 + ((size_t)bh * L_ + qbase) * DH_;
    const _Float16* kptr = k16 + (size_t)bh * L_ * DH_;
    const int r = tid >> 1, c = (tid & 1) * 32;

    auto stageK = [&](int buf, int ch) {   // 4 x 16B per thread
        const _Float16* src = kptr + (size_t)(ch * 128 + r) * DH_ + c;
        _Float16* dst = &Ks[buf][r * 72 + c];
        copy16B(dst,      src);
        copy16B(dst + 8,  src + 8);
        copy16B(dst + 16, src + 16);
        copy16B(dst + 24, src + 24);
    };
    {   // stage the Q tile once
        const _Float16* src = qptr + (size_t)r * DH_ + c;
        _Float16* dst = &Qs[r * 72 + c];
        copy16B(dst,      src);
        copy16B(dst + 8,  src + 8);
        copy16B(dst + 16, src + 16);
        copy16B(dst + 24, src + 24);
    }
    stageK(0, 0);

    float rmx[2][8], rsm[2][8];
#pragma unroll
    for (int ms = 0; ms < 2; ms++)
#pragma unroll
        for (int i = 0; i < 8; i++) { rmx[ms][i] = -3.4e38f; rsm[ms][i] = 0.f; }

    const int nch = L_ / 128;
    for (int ch = 0; ch < nch; ch++) {
        const int cur = ch & 1;
        if (ch + 1 < nch) { stageK(cur ^ 1, ch + 1); wait_copies_keep4(); }
        else              { wait_copies_all(); }
        __syncthreads();

        v8f acc[2][4] = {};
#pragma unroll
        for (int ks = 0; ks < 2; ks++) {   // K = 64 -> two x32 steps
            HF16 af[2], bf[4];
#pragma unroll
            for (int ms = 0; ms < 2; ms++) {
                const _Float16* ap = &Qs[(wm * 32 + ms * 16 + l16) * 72 + ks * 32 + hsel * 8];
                af[ms].h[0] = *(const v8h*)ap;
                af[ms].h[1] = *(const v8h*)(ap + 16);
            }
#pragma unroll
            for (int ns = 0; ns < 4; ns++) {
                const _Float16* bp = &Ks[cur][(wn * 64 + ns * 16 + l16) * 72 + ks * 32 + hsel * 8];
                bf[ns].h[0] = *(const v8h*)bp;
                bf[ns].h[1] = *(const v8h*)(bp + 16);
            }
#pragma unroll
            for (int ms = 0; ms < 2; ms++)
#pragma unroll
                for (int ns = 0; ns < 4; ns++)
                    acc[ms][ns] = __builtin_amdgcn_wmma_f32_16x16x32_f16(
                        false, af[ms].v, false, bf[ns].v, (short)0, acc[ms][ns], false, false);
        }
        // fold this 128-col chunk into running row stats (16-lane butterflies)
#pragma unroll
        for (int ms = 0; ms < 2; ms++) {
#pragma unroll
            for (int i = 0; i < 8; i++) {
                float mxAll = -3.4e38f, smAll = 0.f;
#pragma unroll
                for (int ns = 0; ns < 4; ns++) {
                    float v = acc[ms][ns][i] * SCALE_;
                    float mx = v, sm = v;
#pragma unroll
                    for (int off = 1; off < 16; off <<= 1) {
                        mx = fmaxf(mx, __shfl_xor(mx, off, 16));
                        sm += __shfl_xor(sm, off, 16);
                    }
                    mxAll = fmaxf(mxAll, mx);
                    smAll += sm;
                }
                rmx[ms][i] = fmaxf(rmx[ms][i], mxAll);
                rsm[ms][i] += smAll;
            }
        }
        __syncthreads();
    }
    if (l16 == 0) {   // lanes 0 and 16 publish rows i and i+8
#pragma unroll
        for (int ms = 0; ms < 2; ms++)
#pragma unroll
            for (int i = 0; i < 8; i++) {
                int rr = wm * 32 + ms * 16 + hsel * 8 + i;
                sMax[rr][wn] = rmx[ms][i];
                sSum[rr][wn] = rsm[ms][i];
            }
    }
    __syncthreads();
    if (tid < 128) {
        float mx = fmaxf(sMax[tid][0], sMax[tid][1]);
        float sm = sSum[tid][0] + sSum[tid][1];
        Mstat[(size_t)bh * L_ + qbase + tid] = mx - sm * (1.0f / L_);
    }
}

// ---------------------------------------------------------------------------
// Top-U selection per (b,h): iterative argmax, ties -> lower index (jax top_k)
// ---------------------------------------------------------------------------
__global__ __launch_bounds__(256) void topk_kernel(
    const float* __restrict__ Mstat, int* __restrict__ idxA, int* __restrict__ pos)
{
    __shared__ float mv[L_];
    __shared__ float bmx[256];
    __shared__ int   bix[256];
    int bh = blockIdx.x, tid = threadIdx.x;
    for (int l = tid; l < L_; l += 256) {
        mv[l] = Mstat[(size_t)bh * L_ + l];
        pos[(size_t)bh * L_ + l] = -1;
    }
    __syncthreads();
    for (int u = 0; u < U_; u++) {
        float best = -3.4e38f; int bi = 0;
        for (int l = tid; l < L_; l += 256) {
            float v = mv[l];
            if (v > best) { best = v; bi = l; }
        }
        bmx[tid] = best; bix[tid] = bi;
        __syncthreads();
        for (int s = 128; s > 0; s >>= 1) {
            if (tid < s) {
                float v2 = bmx[tid + s]; int i2 = bix[tid + s];
                if (v2 > bmx[tid] || (v2 == bmx[tid] && i2 < bix[tid])) { bmx[tid] = v2; bix[tid] = i2; }
            }
            __syncthreads();
        }
        if (tid == 0) {
            int w = bix[0];
            idxA[bh * U_ + u] = w;
            pos[(size_t)bh * L_ + w] = u;
            mv[w] = -3.4e38f;
        }
        __syncthreads();
    }
}

// ---------------------------------------------------------------------------
// mean over keys of V per (b,h,d)
// ---------------------------------------------------------------------------
__global__ __launch_bounds__(256) void meanv_kernel(
    const _Float16* __restrict__ v16, float* __restrict__ meanv)
{
    __shared__ float red[256];
    int bh = blockIdx.x, tid = threadIdx.x;
    int d = tid & 63, ck = tid >> 6;
    const _Float16* vp = v16 + (size_t)bh * L_ * DH_;
    float s = 0.f;
    for (int l = ck * 512; l < (ck + 1) * 512; l++) s += (float)vp[(size_t)l * DH_ + d];
    red[tid] = s;
    __syncthreads();
    if (tid < 64) {
        float t = red[tid] + red[tid + 64] + red[tid + 128] + red[tid + 192];
        meanv[bh * DH_ + tid] = t * (1.0f / L_);
    }
}

// ---------------------------------------------------------------------------
// Attention for the U active queries: softmax(q_idx . K^T * scale) @ V
// ---------------------------------------------------------------------------
__global__ __launch_bounds__(256) void attn_kernel(
    const _Float16* __restrict__ q16, const _Float16* __restrict__ k16,
    const _Float16* __restrict__ v16, const int* __restrict__ idxA,
    float* __restrict__ upd)
{
    __shared__ float qrow[DH_];
    __shared__ float p[L_];
    __shared__ float red[256];
    __shared__ float sM, sS;
    int bhu = blockIdx.x;
    int bh = bhu / U_, u = bhu % U_;
    int tid = threadIdx.x;
    int qi = idxA[bh * U_ + u];
    const _Float16* qp = q16 + ((size_t)bh * L_ + qi) * DH_;
    if (tid < DH_) qrow[tid] = (float)qp[tid];
    __syncthreads();
    const _Float16* kp = k16 + (size_t)bh * L_ * DH_;
    float lm = -3.4e38f;
    for (int l = tid; l < L_; l += 256) {
        float s = 0.f;
        const _Float16* kr = kp + (size_t)l * DH_;
#pragma unroll 8
        for (int d = 0; d < DH_; d++) s += qrow[d] * (float)kr[d];
        s *= SCALE_;
        p[l] = s;
        lm = fmaxf(lm, s);
    }
    red[tid] = lm; __syncthreads();
    for (int s2 = 128; s2 > 0; s2 >>= 1) { if (tid < s2) red[tid] = fmaxf(red[tid], red[tid + s2]); __syncthreads(); }
    if (tid == 0) sM = red[0];
    __syncthreads();
    float ls = 0.f;
    for (int l = tid; l < L_; l += 256) { float e = __expf(p[l] - sM); p[l] = e; ls += e; }
    red[tid] = ls; __syncthreads();
    for (int s2 = 128; s2 > 0; s2 >>= 1) { if (tid < s2) red[tid] += red[tid + s2]; __syncthreads(); }
    if (tid == 0) sS = red[0];
    __syncthreads();
    int d = tid & 63, ck = tid >> 6;
    const _Float16* vp = v16 + (size_t)bh * L_ * DH_;
    float acc = 0.f;
    for (int l = ck * 512; l < (ck + 1) * 512; l++) acc += p[l] * (float)vp[(size_t)l * DH_ + d];
    red[tid] = acc; __syncthreads();
    if (tid < 64) {
        float t = red[tid] + red[tid + 64] + red[tid + 128] + red[tid + 192];
        upd[((size_t)bh * U_ + u) * DH_ + tid] = t / sS;
    }
}

// ---------------------------------------------------------------------------
// Assemble context [B,L,D] fp32: mean(V) everywhere, scattered attn rows
// ---------------------------------------------------------------------------
__global__ void ctx_kernel(const int* __restrict__ pos, const float* __restrict__ upd,
                           const float* __restrict__ meanv, float* __restrict__ ctx)
{
    size_t gid = (size_t)blockIdx.x * blockDim.x + threadIdx.x;
    if (gid >= (size_t)B_ * L_ * D_) return;
    int col = gid % D_;
    size_t m = gid / D_;
    int l = m & (L_ - 1);
    int b = m >> 11;
    int hh = col >> 6, dh = col & 63;
    int bh = b * H_ + hh;
    int pp = pos[(size_t)bh * L_ + l];
    ctx[gid] = (pp >= 0) ? upd[((size_t)bh * U_ + pp) * DH_ + dh] : meanv[bh * DH_ + dh];
}

// ---------------------------------------------------------------------------
// h = LayerNorm(h + res) * g + b   (one block per row, in-place safe)
// ---------------------------------------------------------------------------
__global__ __launch_bounds__(256) void add_ln_kernel(
    const float* __restrict__ hin, const float* __restrict__ res,
    const float* __restrict__ g, const float* __restrict__ bta,
    float* __restrict__ hout)
{
    __shared__ float s1[256], s2[256];
    __shared__ float mS, rS;
    int r = blockIdx.x, tid = threadIdx.x;
    const float* hp = hin + (size_t)r * D_;
    const float* ap = res + (size_t)r * D_;
    float x0 = hp[tid] + ap[tid];
    float x1 = hp[tid + 256] + ap[tid + 256];
    s1[tid] = x0 + x1; s2[tid] = x0 * x0 + x1 * x1;
    __syncthreads();
    for (int s = 128; s > 0; s >>= 1) { if (tid < s) { s1[tid] += s1[tid + s]; s2[tid] += s2[tid + s]; } __syncthreads(); }
    if (tid == 0) {
        float mean = s1[0] * (1.0f / D_);
        float var  = s2[0] * (1.0f / D_) - mean * mean;
        mS = mean; rS = rsqrtf(var + 1e-5f);
    }
    __syncthreads();
    hout[(size_t)r * D_ + tid]       = (x0 - mS) * rS * g[tid] + bta[tid];
    hout[(size_t)r * D_ + tid + 256] = (x1 - mS) * rS * g[tid + 256] + bta[tid + 256];
}

// ---------------------------------------------------------------------------
// Output projection: out[b,p,e] = h[b, L-P+p, :] . proj_w[:, e] + proj_b[e]
// ---------------------------------------------------------------------------
__global__ void proj_kernel(const float* __restrict__ h, const float* __restrict__ w,
                            const float* __restrict__ b, float* __restrict__ out)
{
    int gid = blockIdx.x * blockDim.x + threadIdx.x;
    if (gid >= B_ * P_ * E_) return;
    int e = gid % E_;
    int bp = gid / E_;
    int pp = bp % P_, bb = bp / P_;
    size_t row = (size_t)bb * L_ + (L_ - P_) + pp;
    float s = b[e];
    for (int d = 0; d < D_; d++) s += h[row * D_ + d] * w[(size_t)d * E_ + e];
    out[gid] = s;
}

// ---------------------------------------------------------------------------
extern "C" void kernel_launch(void* const* d_in, const int* in_sizes, int n_in,
                              void* d_out, int out_size, void* d_ws, size_t ws_size,
                              hipStream_t stream)
{
    (void)in_sizes; (void)n_in; (void)out_size; (void)ws_size;
    const float* x      = (const float*)d_in[0];
    const float* emb_w  = (const float*)d_in[1];
    const float* emb_b  = (const float*)d_in[2];
    const float* Wq     = (const float*)d_in[3];
    const float* bq     = (const float*)d_in[4];
    const float* Wk     = (const float*)d_in[5];
    const float* bk     = (const float*)d_in[6];
    const float* Wv     = (const float*)d_in[7];
    const float* bv     = (const float*)d_in[8];
    const float* Wo     = (const float*)d_in[9];
    const float* bo     = (const float*)d_in[10];
    const float* W1     = (const float*)d_in[11];
    const float* b1     = (const float*)d_in[12];
    const float* W2     = (const float*)d_in[13];
    const float* b2     = (const float*)d_in[14];
    const float* ln1_g  = (const float*)d_in[15];
    const float* ln1_b  = (const float*)d_in[16];
    const float* ln2_g  = (const float*)d_in[17];
    const float* ln2_b  = (const float*)d_in[18];
    const float* proj_w = (const float*)d_in[19];
    const float* proj_b = (const float*)d_in[20];
    float* out = (float*)d_out;

    char* ws = (char*)d_ws;
    size_t off = 0;
    auto carve = [&](size_t bytes) -> char* {
        char* pp = ws + off;
        off += (bytes + 255) & ~(size_t)255;
        return pp;
    };
    const size_t DD  = (size_t)D_ * D_;
    const size_t BLD = (size_t)B_ * L_ * D_;
    float*    h     = (float*)carve(BLD * 4);
    float*    aB    = (float*)carve(BLD * 4);
    float*    tB    = (float*)carve(BLD * 4);
    _Float16* act16 = (_Float16*)carve(BLD * 2);      // f16 GEMM A operand
    _Float16* q16   = (_Float16*)carve(BLD * 2);
    _Float16* k16   = (_Float16*)carve(BLD * 2);
    _Float16* v16   = (_Float16*)carve(BLD * 2);
    _Float16* w16   = (_Float16*)carve((size_t)NL_ * 6 * DD * 2);
    float*    Mstat = (float*)carve((size_t)B_ * H_ * L_ * 4);
    int*      idxA  = (int*)carve((size_t)B_ * H_ * U_ * 4);
    int*      pos   = (int*)carve((size_t)B_ * H_ * L_ * 4);
    float*    meanv = (float*)carve((size_t)B_ * H_ * DH_ * 4);
    float*    upd   = (float*)carve((size_t)B_ * H_ * U_ * DH_ * 4);

    // Pre-convert all large weights to f16 (col-major B layout)
    for (int lyr = 0; lyr < NL_; lyr++) {
        const float* srcs[6] = { Wq + lyr * DD, Wk + lyr * DD, Wv + lyr * DD,
                                 Wo + lyr * DD, W1 + lyr * DD, W2 + lyr * DD };
        for (int j = 0; j < 6; j++)
            wcvt_kernel<<<(int)(DD / 256), 256, 0, stream>>>(srcs[j], w16 + (lyr * 6 + j) * DD);
    }

    embed_kernel<<<(B_ * L_ * D_) / 256, 256, 0, stream>>>(x, emb_w, emb_b, h);

    const int cvtGrid = (int)(BLD / (256 * 4));
    dim3 gGemm(D_ / 128, (B_ * L_) / 128);   // (4, 64)
    for (int lyr = 0; lyr < NL_; lyr++) {
        const _Float16* wq16 = w16 + (lyr * 6 + 0) * DD;
        const _Float16* wk16 = w16 + (lyr * 6 + 1) * DD;
        const _Float16* wv16 = w16 + (lyr * 6 + 2) * DD;
        const _Float16* wo16 = w16 + (lyr * 6 + 3) * DD;
        const _Float16* w116 = w16 + (lyr * 6 + 4) * DD;
        const _Float16* w216 = w16 + (lyr * 6 + 5) * DD;

        cvt16_kernel<<<cvtGrid, 256, 0, stream>>>(h, act16);
        gemm_f16<1, 0><<<gGemm, 256, 0, stream>>>(act16, wq16, bq + lyr * D_, nullptr, q16,
                                                  B_ * L_, D_, D_);
        gemm_f16<1, 0><<<gGemm, 256, 0, stream>>>(act16, wk16, bk + lyr * D_, nullptr, k16,
                                                  B_ * L_, D_, D_);
        gemm_f16<1, 0><<<gGemm, 256, 0, stream>>>(act16, wv16, bv + lyr * D_, nullptr, v16,
                                                  B_ * L_, D_, D_);

        score_stats<<<dim3(L_ / 128, B_ * H_), 256, 0, stream>>>(q16, k16, Mstat);
        topk_kernel<<<B_ * H_, 256, 0, stream>>>(Mstat, idxA, pos);
        meanv_kernel<<<B_ * H_, 256, 0, stream>>>(v16, meanv);
        attn_kernel<<<B_ * H_ * U_, 256, 0, stream>>>(q16, k16, v16, idxA, upd);
        ctx_kernel<<<(B_ * L_ * D_) / 256, 256, 0, stream>>>(pos, upd, meanv, tB);

        cvt16_kernel<<<cvtGrid, 256, 0, stream>>>(tB, act16);
        gemm_f16<0, 0><<<gGemm, 256, 0, stream>>>(act16, wo16, bo + lyr * D_, aB, nullptr,
                                                  B_ * L_, D_, D_);
        add_ln_kernel<<<B_ * L_, 256, 0, stream>>>(h, aB, ln1_g + lyr * D_, ln1_b + lyr * D_, h);

        cvt16_kernel<<<cvtGrid, 256, 0, stream>>>(h, act16);
        gemm_f16<0, 1><<<gGemm, 256, 0, stream>>>(act16, w116, b1 + lyr * D_, tB, nullptr,
                                                  B_ * L_, D_, D_);
        cvt16_kernel<<<cvtGrid, 256, 0, stream>>>(tB, act16);
        gemm_f16<0, 0><<<gGemm, 256, 0, stream>>>(act16, w216, b2 + lyr * D_, aB, nullptr,
                                                  B_ * L_, D_, D_);
        add_ln_kernel<<<B_ * L_, 256, 0, stream>>>(h, aB, ln2_g + lyr * D_, ln2_b + lyr * D_, h);
    }

    proj_kernel<<<(B_ * P_ * E_ + 255) / 256, 256, 0, stream>>>(h, proj_w, proj_b, out);
}
